// RAFT_19327352832742
// MI455X (gfx1250) — compile-verified, matrix operand order checked by reading, not run on hardware
//
#include <hip/hip_runtime.h>
#include <hip/hip_bf16.h>

typedef __bf16 bf16_t;
typedef __attribute__((ext_vector_type(16))) __bf16 v16bf;
typedef __attribute__((ext_vector_type(8)))  float  v8f;
typedef __attribute__((ext_vector_type(4)))  unsigned int u32x4;
struct U32x8 { u32x4 lo, hi; };

#define NPOLY 256
#define PLEN  128
#define HH    128
#define WW    128
#define CF    64

#define CSTAGE 128             // channels staged in LDS per pass
#define HALO   16              // max halo: k9, dil4 -> 4*4
#define SROW   (PLEN + 2*HALO) // 160 columns incl. halo
#define CROW   136             // channel-dim stride (272B: 16B-aligned, odd*16 -> bank spread)

__device__ __forceinline__ unsigned short f2bfu(float f) {
    unsigned u = __builtin_bit_cast(unsigned, f);
    unsigned r = u + 0x7FFFu + ((u >> 16) & 1u);
    return (unsigned short)(r >> 16);
}
__device__ __forceinline__ bf16_t u2bf(unsigned short h) {
    return __builtin_bit_cast(bf16_t, h);
}
__device__ __forceinline__ float act_f(float v, int act) {
    if (act == 1) return fmaxf(v, 0.f);
    if (act == 2) return 1.f / (1.f + __expf(-v));
    if (act == 3) return tanhf(v);
    return v;
}

// ---------------------------------------------------------------------------
// Repack conv weights (fp32 [Cout, Ctot*KS], layout w[m, ci*KS+kk]) into
// WMMA A-fragment order with TAP-MAJOR K enumeration:
//   block b = (ct*KS + kk)*nkc + cblk  covers K = 32 consecutive channels at
//   fixed tap kk;  dst[(b*32 + lane)*16 + 2j+t] = w[m, ci*KS+kk],
//   ci = cblk*32 + (j/4)*16 + (lane/16)*8 + (j%4)*2 + t,  m = ct*16 + lane%16.
// Out-of-range (channel-padding) entries are zero.
// ---------------------------------------------------------------------------
__global__ void repack_w_kernel(const float* __restrict__ w, bf16_t* __restrict__ dst,
                                int Cout, int Ctot, int KS, int nkc, long total)
{
    const long i = (long)blockIdx.x * blockDim.x + threadIdx.x;
    if (i >= total) return;
    const int  e    = (int)(i & 15);
    const int  lane = (int)((i >> 4) & 31);
    long blk = i >> 9;
    const int cblk = (int)(blk % nkc); blk /= nkc;
    const int kk   = (int)(blk % KS);
    const int ct   = (int)(blk / KS);
    const int j = e >> 1, t = e & 1;
    const int ci = (cblk << 5) + ((j >> 2) << 4) + ((lane >> 4) << 3) + ((j & 3) << 1) + t;
    const int m  = (ct << 4) + (lane & 15);
    float v = 0.f;
    if (m < Cout && ci < Ctot) v = w[(long)m * Ctot * KS + (long)ci * KS + kk];
    dst[i] = u2bf(f2bfu(v));
}

// ---------------------------------------------------------------------------
// 1-D conv as GEMM on WMMA bf16.  Block = 128 thr (4 waves) = one (n, ct).
// Wave w owns column tiles p0 = 32w, p1 = 32w+16; one A fragment feeds 2 WMMAs.
// Activations staged TRANSPOSED in LDS ([col][channel], halo materialized,
// channel blocks zero-padded to 32); with tap-major K order each B fragment
// is two contiguous 8-channel runs -> 2x ds_load_b128 per fragment.
// ---------------------------------------------------------------------------
template <int KS, int CIRC>
__global__ void conv_wmma_kernel(const bf16_t* __restrict__ wpk,
                                 const float*  __restrict__ bias,
                                 const float*  __restrict__ x, long x_sstride,
                                 const float*  __restrict__ gvec, int g_ch,
                                 float* __restrict__ y, long y_sstride,
                                 int Cin, int Cout, int dil, int act, int nkc)
{
    __shared__ unsigned short sh[SROW * CROW];   // [col][channel]

    const int tid  = threadIdx.x;
    const int wave = tid >> 5;
    const int lane = tid & 31;
    const int l15  = lane & 15;
    const int half = lane >> 4;
    const int ct   = blockIdx.x;
    const int n    = blockIdx.y;

    const int Ctot   = g_ch + Cin;
    const int center = (KS - 1) >> 1;
    const float* xn  = x + (long)n * x_sstride;
    const int p0 = wave << 5;

    v8f acc0 = {0.f,0.f,0.f,0.f,0.f,0.f,0.f,0.f};
    v8f acc1 = {0.f,0.f,0.f,0.f,0.f,0.f,0.f,0.f};

    for (int cb = 0; cb < Ctot; cb += CSTAGE) {
        const int rows    = min(CSTAGE, Ctot - cb);
        const int rowsPad = (rows + 31) & ~31;
        // ---- cooperative transposed stage (halo + zero-padding resolved) ----
        for (int idx = tid; idx < rowsPad * SROW; idx += 128) {
            const int rl  = idx / SROW;
            const int col = idx - rl * SROW;
            const int ci  = cb + rl;
            float v = 0.f;
            if (ci < g_ch) {
                v = gvec[(long)n * g_ch + ci];
            } else {
                const int cc = ci - g_ch;
                int q = col - HALO;
                if (CIRC) q &= (PLEN - 1);
                if (cc < Cin && q >= 0 && q < PLEN) v = xn[(long)cc * PLEN + q];
            }
            sh[col * CROW + rl] = f2bfu(v);
        }
        __syncthreads();

        const int cblk0 = cb >> 5;
        const int cblkN = (cb + rowsPad) >> 5;
        for (int kk = 0; kk < KS; ++kk) {
            const int col0 = HALO + (kk - center) * dil + p0 + l15;
            const unsigned short* colp0 = sh + (long)col0 * CROW;
            const unsigned short* colp1 = colp0 + 16 * CROW;     // second p-tile
            for (int cblk = cblk0; cblk < cblkN; ++cblk) {
                const long ablk = ((long)((ct * KS + kk) * nkc + cblk) * 32 + lane) * 16;
                __builtin_prefetch(wpk + ablk + 512, 0, 1);      // next A block
                const v16bf af = *(const v16bf*)(wpk + ablk);    // 2x global b128

                const int rbase = ((cblk << 5) - cb) + (half << 3);
                const u32x4 b0lo = *(const u32x4*)(colp0 + rbase);       // ds b128
                const u32x4 b0hi = *(const u32x4*)(colp0 + rbase + 16);  // ds b128
                const u32x4 b1lo = *(const u32x4*)(colp1 + rbase);
                const u32x4 b1hi = *(const u32x4*)(colp1 + rbase + 16);
                const v16bf b0 = __builtin_bit_cast(v16bf, U32x8{b0lo, b0hi});
                const v16bf b1 = __builtin_bit_cast(v16bf, U32x8{b1lo, b1hi});

                acc0 = __builtin_amdgcn_wmma_f32_16x16x32_bf16(
                           false, af, false, b0, (short)0, acc0, false, false);
                acc1 = __builtin_amdgcn_wmma_f32_16x16x32_bf16(
                           false, af, false, b1, (short)0, acc1, false, false);
            }
        }
        __syncthreads();
    }

#pragma unroll
    for (int r = 0; r < 8; ++r) {
        const int mo = (ct << 4) + r + (half << 3);
        if (mo < Cout) {
            const float b = bias[mo];
            const long yb = (long)n * y_sstride + (long)mo * PLEN + l15;
            y[yb + p0]      = act_f(acc0[r] + b, act);
            y[yb + p0 + 16] = act_f(acc1[r] + b, act);
        }
    }
}

// ---------------------------------------------------------------------------
// Bilinear gather of cnn_feature at poly points + canonical-poly channels.
// ---------------------------------------------------------------------------
__global__ void gather_x66_kernel(const float* __restrict__ cnn,
                                  const float* __restrict__ poly,
                                  const int*   __restrict__ ind,
                                  float* __restrict__ x66)
{
    const int n = blockIdx.x;
    const int p = threadIdx.x;
    __shared__ float smx[PLEN], smy[PLEN];
    const float px = poly[((long)n * PLEN + p) * 2 + 0];
    const float py = poly[((long)n * PLEN + p) * 2 + 1];
    smx[p] = px; smy[p] = py;
    __syncthreads();
    for (int s = PLEN >> 1; s > 0; s >>= 1) {
        if (p < s) { smx[p] = fminf(smx[p], smx[p + s]);
                     smy[p] = fminf(smy[p], smy[p + s]); }
        __syncthreads();
    }
    const float mnx = smx[0], mny = smy[0];
    float* xr = x66 + (long)n * 66 * PLEN;
    xr[64 * PLEN + p] = px - mnx;
    xr[65 * PLEN + p] = py - mny;

    const float fx = px * WW - 0.5f;
    const float fy = py * HH - 0.5f;
    const int   x0 = (int)floorf(fx);
    const int   y0 = (int)floorf(fy);
    const float wx = fx - (float)x0;
    const float wy = fy - (float)y0;
    const float* fb = cnn + (long)ind[n] * CF * HH * WW;

    int   xs[4]  = {x0, x0 + 1, x0, x0 + 1};
    int   ysv[4] = {y0, y0, y0 + 1, y0 + 1};
    float wv[4]  = {(1.f - wx) * (1.f - wy), wx * (1.f - wy),
                    (1.f - wx) * wy,         wx * wy};
    long  idxv[4];
#pragma unroll
    for (int t = 0; t < 4; ++t) {
        bool valid = (xs[t] >= 0) && (xs[t] < WW) && (ysv[t] >= 0) && (ysv[t] < HH);
        int xc = min(max(xs[t], 0), WW - 1);
        int yc = min(max(ysv[t], 0), HH - 1);
        idxv[t] = (long)yc * WW + xc;
        if (!valid) wv[t] = 0.f;
    }
    for (int c = 0; c < CF; ++c) {
        const float* fc = fb + (long)c * HH * WW;
        float v = fc[idxv[0]] * wv[0] + fc[idxv[1]] * wv[1]
                + fc[idxv[2]] * wv[2] + fc[idxv[3]] * wv[3];
        xr[(long)c * PLEN + p] = v;
    }
}

// ---------------- batch-norm (two pass) ------------------------------------
__global__ void bn_stats_kernel(const float* __restrict__ x, float* __restrict__ stats)
{
    const int c = blockIdx.x;
    const int t = threadIdx.x;
    float s = 0.f, s2 = 0.f;
    for (int i = t; i < NPOLY * PLEN; i += 256) {
        const int n = i >> 7, p = i & 127;
        const float v = x[((long)n * 128 + c) * PLEN + p];
        s += v; s2 += v * v;
    }
    __shared__ float sh[256], sh2[256];
    sh[t] = s; sh2[t] = s2; __syncthreads();
    for (int st = 128; st > 0; st >>= 1) {
        if (t < st) { sh[t] += sh[t + st]; sh2[t] += sh2[t + st]; }
        __syncthreads();
    }
    if (t == 0) {
        const float inv = 1.f / (float)(NPOLY * PLEN);
        const float mean = sh[0] * inv;
        const float var  = sh2[0] * inv - mean * mean;
        stats[c * 2 + 0] = mean;
        stats[c * 2 + 1] = rsqrtf(var + 1e-5f);
    }
}

__global__ void bn_apply_kernel(const float* __restrict__ x,
                                const float* __restrict__ stats,
                                const float* __restrict__ gam,
                                const float* __restrict__ bet,
                                const float* __restrict__ res, long res_sstride,
                                float* __restrict__ out, long out_sstride)
{
    const long i = (long)blockIdx.x * blockDim.x + threadIdx.x;
    if (i >= (long)NPOLY * 128 * PLEN) return;
    const int p = i & 127;
    const int c = (int)((i >> 7) & 127);
    const int n = (int)(i >> 14);
    float v = x[((long)n * 128 + c) * PLEN + p];
    v = (v - stats[c * 2]) * stats[c * 2 + 1] * gam[c] + bet[c];
    if (res) v += res[(long)n * res_sstride + (long)c * PLEN + p];
    out[(long)n * out_sstride + (long)c * PLEN + p] = v;
}

// ---------------- small elementwise / reduction helpers ---------------------
__global__ void max_over_p_kernel(const float* __restrict__ y, float* __restrict__ g, int C)
{
    const int idx = blockIdx.x * blockDim.x + threadIdx.x;
    if (idx >= NPOLY * C) return;
    const int n = idx / C, c = idx % C;
    const float* row = y + ((long)n * C + c) * PLEN;
    float m = row[0];
    for (int p = 1; p < PLEN; ++p) m = fmaxf(m, row[p]);
    g[idx] = m;
}

__global__ void tanh_leaky_kernel(const float* __restrict__ in,
                                  float* __restrict__ net, float* __restrict__ fea)
{
    const long i = (long)blockIdx.x * blockDim.x + threadIdx.x;
    if (i >= (long)NPOLY * 128 * PLEN) return;
    const float v = in[i];
    net[i] = tanhf(v);
    fea[i] = (v > 0.f) ? v : 0.01f * v;
}

__global__ void leaky_kernel(const float* __restrict__ in, float* __restrict__ fea)
{
    const long i = (long)blockIdx.x * blockDim.x + threadIdx.x;
    if (i >= (long)NPOLY * 128 * PLEN) return;
    const float v = in[i];
    fea[i] = (v > 0.f) ? v : 0.01f * v;
}

__global__ void concat2_kernel(const float* __restrict__ a, const float* __restrict__ b,
                               float* __restrict__ o)
{
    const long i = (long)blockIdx.x * blockDim.x + threadIdx.x;
    if (i >= (long)NPOLY * 256 * PLEN) return;
    const int p = i & 127;
    const int c = (int)((i >> 7) & 255);
    const int n = (int)(i >> 15);
    o[i] = (c < 128) ? a[((long)n * 128 + c) * PLEN + p]
                     : b[((long)n * 128 + (c - 128)) * PLEN + p];
}

__global__ void rmul_concat_kernel(const float* __restrict__ r, const float* __restrict__ net,
                                   const float* __restrict__ fea, float* __restrict__ o)
{
    const long i = (long)blockIdx.x * blockDim.x + threadIdx.x;
    if (i >= (long)NPOLY * 256 * PLEN) return;
    const int p = i & 127;
    const int c = (int)((i >> 7) & 255);
    const int n = (int)(i >> 15);
    if (c < 128) {
        const long j = ((long)n * 128 + c) * PLEN + p;
        o[i] = r[j] * net[j];
    } else {
        o[i] = fea[((long)n * 128 + (c - 128)) * PLEN + p];
    }
}

__global__ void gate_kernel(const float* __restrict__ z, const float* __restrict__ q,
                            float* __restrict__ net)
{
    const long i = (long)blockIdx.x * blockDim.x + threadIdx.x;
    if (i >= (long)NPOLY * 128 * PLEN) return;
    const float zv = z[i];
    net[i] = (1.f - zv) * net[i] + zv * q[i];
}

__global__ void add_off_kernel(const float* __restrict__ off, float* __restrict__ py,
                               float* __restrict__ outsl)
{
    const long i = (long)blockIdx.x * blockDim.x + threadIdx.x;
    if (i >= (long)NPOLY * PLEN * 2) return;
    const int d = (int)(i & 1);
    const int p = (int)((i >> 1) & 127);
    const int n = (int)(i >> 8);
    const float v = py[i] + off[((long)n * 2 + d) * PLEN + p];
    py[i] = v;
    outsl[i] = v;
}

__global__ void copy_kernel(const float* __restrict__ a, float* __restrict__ b, long n)
{
    const long i = (long)blockIdx.x * blockDim.x + threadIdx.x;
    if (i < n) b[i] = a[i];
}

// ---------------------------------------------------------------------------
extern "C" void kernel_launch(void* const* d_in, const int* in_sizes, int n_in,
                              void* d_out, int out_size, void* d_ws, size_t ws_size,
                              hipStream_t stream)
{
    (void)in_sizes; (void)n_in; (void)out_size; (void)ws_size;

    // ---- inputs (jax tree order: dicts sorted by key) ----
    const float* cnn   = (const float*)d_in[0];
    const float* poly  = (const float*)d_in[1];
    const int*   ind   = (const int*)  d_in[2];
    const float* fus_b = (const float*)d_in[3];
    const float* fus_w = (const float*)d_in[4];
    const float* hd_b  = (const float*)d_in[5];
    const float* hd_be = (const float*)d_in[6];
    const float* hd_g  = (const float*)d_in[7];
    const float* hd_w  = (const float*)d_in[8];
    const float* p0_b  = (const float*)d_in[9];
    const float* p0_w  = (const float*)d_in[10];
    const float* p1_b  = (const float*)d_in[11];
    const float* p1_w  = (const float*)d_in[12];
    const float *rs_b[7], *rs_be[7], *rs_g[7], *rs_w[7];
    for (int i = 0; i < 7; ++i) {
        rs_b[i]  = (const float*)d_in[13 + 4 * i];
        rs_be[i] = (const float*)d_in[14 + 4 * i];
        rs_g[i]  = (const float*)d_in[15 + 4 * i];
        rs_w[i]  = (const float*)d_in[16 + 4 * i];
    }
    const float* q_b   = (const float*)d_in[41];
    const float* q_w   = (const float*)d_in[42];
    const float* r_b   = (const float*)d_in[43];
    const float* r_w   = (const float*)d_in[44];
    const float* z_b   = (const float*)d_in[45];
    const float* z_w   = (const float*)d_in[46];
    const float* up0_b = (const float*)d_in[47];
    const float* up0_w = (const float*)d_in[48];
    const float* up1_b = (const float*)d_in[49];
    const float* up1_w = (const float*)d_in[50];
    const float* up2_b = (const float*)d_in[51];
    const float* up2_w = (const float*)d_in[52];

    // ---- workspace layout (floats) ----
    float* f = (float*)d_ws;
    long off = 0;
    auto alloc = [&](long nf) { float* p = f + off; off += nf; return p; };
    float* X66    = alloc((long)NPOLY * 66 * PLEN);
    float* STATES = alloc((long)NPOLY * 1024 * PLEN);
    float* TMP1   = alloc((long)NPOLY * 256 * PLEN);
    float* TMP2   = alloc((long)NPOLY * 256 * PLEN);
    float* NET    = alloc((long)NPOLY * 128 * PLEN);
    float* FEA    = alloc((long)NPOLY * 128 * PLEN);
    float* SNOUT  = alloc((long)NPOLY * 128 * PLEN);
    float* HX     = alloc((long)NPOLY * 256 * PLEN);
    float* Zb     = alloc((long)NPOLY * 128 * PLEN);
    float* Rb     = alloc((long)NPOLY * 128 * PLEN);
    float* Qb     = alloc((long)NPOLY * 128 * PLEN);
    float* G      = alloc((long)NPOLY * 256);
    float* STATS  = alloc(256);
    float* PY     = alloc((long)NPOLY * PLEN * 2);
    off = (off + 7) & ~7L;                       // 32B-align packed weights

    bf16_t* wb = (bf16_t*)(f + off);
    long wo = 0;

    struct PackedW { const bf16_t* pk; int nkc; };
    auto repack = [&](const float* src, int Cout, int Ctot, int KS) -> PackedW {
        const int nct = (Cout + 15) / 16;
        const int nkc = (Ctot + 31) / 32;
        bf16_t* dst = wb + wo;
        const long total = (long)nct * KS * nkc * 512;
        wo += total;
        repack_w_kernel<<<dim3((unsigned)((total + 255) / 256)), 256, 0, stream>>>(
            src, dst, Cout, Ctot, KS, nkc, total);
        return {dst, nkc};
    };

    PackedW HDW  = repack(hd_w, 128, 66, 9);
    PackedW RSW[7];
    for (int i = 0; i < 7; ++i) RSW[i] = repack(rs_w[i], 128, 128, 9);
    PackedW FUSW = repack(fus_w, 256, 1024, 1);
    PackedW P0W  = repack(p0_w, 256, 1280, 1);
    PackedW P1W  = repack(p1_w, 128, 256, 1);
    PackedW QW   = repack(q_w, 128, 256, 3);
    PackedW RW   = repack(r_w, 128, 256, 3);
    PackedW ZW   = repack(z_w, 128, 256, 3);
    PackedW U0W  = repack(up0_w, 256, 128, 1);
    PackedW U1W  = repack(up1_w, 64, 256, 1);
    PackedW U2W  = repack(up2_w, 2, 64, 1);

    auto conv = [&](PackedW w, const float* b, const float* x, long xs,
                    const float* g, int gch, float* y, long ys,
                    int Cin, int Cout, int ks, int dil, int act) {
        dim3 grid((Cout + 15) / 16, NPOLY);
        if (ks == 9)
            conv_wmma_kernel<9, 1><<<grid, 128, 0, stream>>>(
                w.pk, b, x, xs, g, gch, y, ys, Cin, Cout, dil, act, w.nkc);
        else if (ks == 3)
            conv_wmma_kernel<3, 0><<<grid, 128, 0, stream>>>(
                w.pk, b, x, xs, g, gch, y, ys, Cin, Cout, dil, act, w.nkc);
        else
            conv_wmma_kernel<1, 0><<<grid, 128, 0, stream>>>(
                w.pk, b, x, xs, g, gch, y, ys, Cin, Cout, dil, act, w.nkc);
    };

    const long EW128 = (long)NPOLY * 128 * PLEN;
    const long EW256 = (long)NPOLY * 256 * PLEN;
    const unsigned G128 = (unsigned)((EW128 + 255) / 256);
    const unsigned G256 = (unsigned)((EW256 + 255) / 256);
    const int DIL[7] = {1, 1, 1, 2, 2, 4, 4};

    auto run_snake = [&]() {
        conv(HDW, hd_b, X66, 66L * PLEN, nullptr, 0, TMP1, 128L * PLEN,
             66, 128, 9, 1, /*relu*/1);
        bn_stats_kernel<<<128, 256, 0, stream>>>(TMP1, STATS);
        bn_apply_kernel<<<G128, 256, 0, stream>>>(TMP1, STATS, hd_g, hd_be,
                                                  nullptr, 0, STATES, 1024L * PLEN);
        for (int i = 0; i < 7; ++i) {
            const float* xin = STATES + (long)i * 128 * PLEN;
            conv(RSW[i], rs_b[i], xin, 1024L * PLEN, nullptr, 0, TMP1, 128L * PLEN,
                 128, 128, 9, DIL[i], 1);
            bn_stats_kernel<<<128, 256, 0, stream>>>(TMP1, STATS);
            bn_apply_kernel<<<G128, 256, 0, stream>>>(TMP1, STATS, rs_g[i], rs_be[i],
                                                      xin, 1024L * PLEN,
                                                      STATES + (long)(i + 1) * 128 * PLEN,
                                                      1024L * PLEN);
        }
        conv(FUSW, fus_b, STATES, 1024L * PLEN, nullptr, 0, TMP1, 256L * PLEN,
             1024, 256, 1, 1, 0);
        max_over_p_kernel<<<(NPOLY * 256 + 255) / 256, 256, 0, stream>>>(TMP1, G, 256);
        conv(P0W, p0_b, STATES, 1024L * PLEN, G, 256, TMP2, 256L * PLEN,
             1024, 256, 1, 1, 1);
        conv(P1W, p1_b, TMP2, 256L * PLEN, nullptr, 0, SNOUT, 128L * PLEN,
             256, 128, 1, 1, 0);
    };

    // ---- forward ----
    copy_kernel<<<(NPOLY * PLEN * 2 + 255) / 256, 256, 0, stream>>>(
        poly, PY, (long)NPOLY * PLEN * 2);
    gather_x66_kernel<<<NPOLY, PLEN, 0, stream>>>(cnn, PY, ind, X66);
    run_snake();
    tanh_leaky_kernel<<<G128, 256, 0, stream>>>(SNOUT, NET, FEA);

    for (int it = 0; it < 6; ++it) {
        concat2_kernel<<<G256, 256, 0, stream>>>(NET, FEA, HX);
        conv(ZW, z_b, HX, 256L * PLEN, nullptr, 0, Zb, 128L * PLEN,
             256, 128, 3, 1, /*sigmoid*/2);
        conv(RW, r_b, HX, 256L * PLEN, nullptr, 0, Rb, 128L * PLEN,
             256, 128, 3, 1, 2);
        rmul_concat_kernel<<<G256, 256, 0, stream>>>(Rb, NET, FEA, HX);
        conv(QW, q_b, HX, 256L * PLEN, nullptr, 0, Qb, 128L * PLEN,
             256, 128, 3, 1, /*tanh*/3);
        gate_kernel<<<G128, 256, 0, stream>>>(Zb, Qb, NET);
        conv(U0W, up0_b, NET, 128L * PLEN, nullptr, 0, TMP1, 256L * PLEN,
             128, 256, 1, 1, 1);
        conv(U1W, up1_b, TMP1, 256L * PLEN, nullptr, 0, TMP2, 64L * PLEN,
             256, 64, 1, 1, 1);
        conv(U2W, up2_b, TMP2, 64L * PLEN, nullptr, 0, Rb /*OFF*/, 2L * PLEN,
             64, 2, 1, 1, 0);
        add_off_kernel<<<(NPOLY * PLEN * 2 + 255) / 256, 256, 0, stream>>>(
            Rb, PY, (float*)d_out + (long)it * NPOLY * PLEN * 2);

        if (it < 5) {
            gather_x66_kernel<<<NPOLY, PLEN, 0, stream>>>(cnn, PY, ind, X66);
            run_snake();
            leaky_kernel<<<G128, 256, 0, stream>>>(SNOUT, FEA);
        }
    }
}
